// e3nnPrior_54142357733962
// MI455X (gfx1250) — matrix operand order repaired and checked by reading, not compiled, not run on hardware
//
#include <hip/hip_runtime.h>
#include <math.h>

#define NNODES 50000
#define EUND   150000
#define NE     300000

typedef __attribute__((ext_vector_type(16))) _Float16 v16h;
typedef __attribute__((ext_vector_type(8)))  _Float16 v8h;
typedef __attribute__((ext_vector_type(8)))  float    v8f;

union V16U { v16h v; v8h h[2]; };

// ---------- WMMA fragment loaders (CDNA5 16-bit A / B layouts, wave32) ----------

static __device__ inline v16h load_a_frag(const _Float16* base, int ld, int lane) {
  // A 16x32 f16: lane<16 -> row=lane, K halves {0..7,16..23}; lane>=16 -> row=lane-16, K {8..15,24..31}
  int row = lane & 15;
  int kb  = (lane >> 4) << 3;
  const _Float16* p = base + row * ld + kb;
  V16U t;
  t.h[0] = *(const v8h*)(p);
  t.h[1] = *(const v8h*)(p + 16);
  return t.v;
}

static __device__ inline v16h load_b_packed(const _Float16* panel, int lane) {
  // panel pre-swizzled into lane-major order: lane*16 contiguous halves
  const _Float16* p = panel + lane * 16;
  V16U t;
  t.h[0] = *(const v8h*)(p);
  t.h[1] = *(const v8h*)(p + 8);
  return t.v;
}

// ---------- pack f32 row-major (K x N) weights into WMMA-B panels of 32x16 ----------
// dst[((kp*NP + np)*32 + lane)*16 + h] ; lane<16: col=np*16+lane, k=kp*32+{0..7,16..23}[h]
//                                         lane>=16: col=np*16+lane-16, k=kp*32+{8..15,24..31}[h]
__global__ __launch_bounds__(256) void pack_b_kernel(const float* __restrict__ W,
                                                     _Float16* __restrict__ dst,
                                                     int K, int N, int KP) {
  int idx = blockIdx.x * blockDim.x + threadIdx.x;
  int NP = N >> 4;
  int total = (KP >> 5) * NP * 512;
  if (idx >= total) return;
  int h     = idx & 15;
  int lane  = (idx >> 4) & 31;
  int panel = idx >> 9;
  int np    = panel % NP;
  int kp    = panel / NP;
  int col = np * 16 + (lane & 15);
  int k   = kp * 32 + ((lane >> 4) << 3) + (h & 7) + ((h >> 3) << 4);
  float v = (k < K) ? W[(size_t)k * N + col] : 0.0f;
  dst[idx] = (_Float16)v;
}

// ---------- edge preparation: geometry, RBF, spherical harmonics, edge MLP, degree ----------

__global__ __launch_bounds__(256) void edge_prep_kernel(
    const int* __restrict__ cg_z, const float* __restrict__ xyz,
    const int* __restrict__ nbr,
    const float* __restrict__ ew1, const float* __restrict__ eb1,
    const float* __restrict__ ew2, const float* __restrict__ eb2,
    float* __restrict__ shbuf, float* __restrict__ eabuf, float* __restrict__ counts)
{
  int e = blockIdx.x * blockDim.x + threadIdx.x;
  if (e >= NE) return;
  int eu = (e < EUND) ? e : e - EUND;
  int n0 = nbr[eu * 2 + 0];
  int n1 = nbr[eu * 2 + 1];
  int src = (e < EUND) ? n0 : n1;
  int dst = (e < EUND) ? n1 : n0;

  float rx = xyz[dst * 3 + 0] - xyz[src * 3 + 0];
  float ry = xyz[dst * 3 + 1] - xyz[src * 3 + 1];
  float rz = xyz[dst * 3 + 2] - xyz[src * 3 + 2];
  float d   = sqrtf(rx * rx + ry * ry + rz * rz);
  float inv = 1.0f / d;
  float x = rx * inv, y = ry * inv, z = rz * inv;

  const float s3 = 1.73205080757f, s15 = 3.87298334621f, s5 = 2.2360679775f;
  float s[9];
  s[0] = 1.0f;
  s[1] = s3 * y;  s[2] = s3 * z;  s[3] = s3 * x;
  s[4] = s15 * x * y;  s[5] = s15 * y * z;
  s[6] = 0.5f * s5 * (3.0f * z * z - 1.0f);
  s[7] = s15 * x * z;
  s[8] = 0.5f * s15 * (x * x - y * y);
#pragma unroll
  for (int i = 0; i < 9; ++i) shbuf[(size_t)e * 9 + i] = s[i];

  float attr[14];
  attr[0] = (float)cg_z[src];
  attr[1] = (float)cg_z[dst];
#pragma unroll
  for (int i = 2; i < 6; ++i) attr[i] = 0.0f;
  const float coeff = -49.0f / 1800.0f;  // -0.5/(30/7)^2
#pragma unroll
  for (int k = 0; k < 8; ++k) {
    float dd = d - (30.0f / 7.0f) * (float)k;
    attr[6 + k] = expf(coeff * dd * dd);
  }

  float hh[12];
#pragma unroll
  for (int i = 0; i < 12; ++i) {
    float v = eb1[i];
#pragma unroll
    for (int j = 0; j < 14; ++j) v += attr[j] * ew1[j * 12 + i];
    hh[i] = fmaxf(v, 0.0f);
  }
#pragma unroll
  for (int i = 0; i < 12; ++i) {
    float v = eb2[i];
#pragma unroll
    for (int j = 0; j < 12; ++j) v += hh[j] * ew2[j * 12 + i];
    eabuf[(size_t)e * 12 + i] = v;
  }
  atomicAdd(&counts[src], 1.0f);
}

// ---------- node init / update / f16 convert ----------

__global__ __launch_bounds__(256) void node_init_kernel(const int* __restrict__ z,
                                                        const float* __restrict__ emb,
                                                        float* __restrict__ node) {
  int idx = blockIdx.x * blockDim.x + threadIdx.x;
  if (idx >= NNODES * 48) return;
  int i = idx / 48, j = idx % 48;
  int zz = z[i];
  node[idx] = (j < 12 && zz != 0) ? emb[zz * 12 + j] : 0.0f;
}

__global__ __launch_bounds__(256) void node_update_kernel(const float* __restrict__ upd,
                                                          const float* __restrict__ counts,
                                                          float* __restrict__ node) {
  int idx = blockIdx.x * blockDim.x + threadIdx.x;
  if (idx >= NNODES * 48) return;
  int i = idx / 48;
  float c = fmaxf(counts[i], 1.0f);
  node[idx] += upd[idx] / c;
}

__global__ __launch_bounds__(256) void node_to_f16_kernel(const float* __restrict__ node,
                                                          _Float16* __restrict__ nf) {
  int idx = blockIdx.x * blockDim.x + threadIdx.x;
  if (idx >= NNODES * 64) return;
  int i = idx >> 6, j = idx & 63;
  nf[idx] = (j < 48) ? (_Float16)node[i * 48 + j] : (_Float16)0.0f;
}

// ---------- per-layer tensor-product path tables (compile-time) ----------
// lt: 0:(l1,l2,l3)=(0,0,0) 1:(0,1,1) 2:(1,0,1) 3:(1,1,0) 4:(1,1,1) 5:(1,2,1)

template <int L> struct LayerDef;
template <> struct LayerDef<0> {
  static constexpr int NP = 2, WN = 192, OUTD = 24;
  static constexpr int   mi[NP]  = {12, 12};
  static constexpr int   mo[NP]  = {12, 4};
  static constexpr int   lt[NP]  = {0, 1};
  static constexpr int   ino[NP] = {0, 0};
  static constexpr int   oo[NP]  = {0, 12};
  static constexpr int   wf[NP]  = {0, 144};
  static constexpr float al[NP]  = {0.288675134f, 0.5f};
};
template <> struct LayerDef<1> {
  static constexpr int NP = 6, WN = 288, OUTD = 36;
  static constexpr int   mi[NP]  = {12, 12, 4, 4, 4, 4};
  static constexpr int   mo[NP]  = {12, 4, 4, 12, 4, 4};
  static constexpr int   lt[NP]  = {0, 1, 2, 3, 4, 5};
  static constexpr int   ino[NP] = {0, 0, 12, 12, 12, 12};
  static constexpr int   oo[NP]  = {0, 12, 12, 0, 24, 12};
  static constexpr int   wf[NP]  = {0, 144, 192, 208, 256, 272};
  static constexpr float al[NP]  = {0.25f, 0.387298335f, 0.387298335f, 0.25f, 0.866025404f, 0.387298335f};
};
template <> struct LayerDef<2> {
  static constexpr int NP = 10, WN = 384, OUTD = 48;
  static constexpr int   mi[NP]  = {12, 12, 4, 4, 4, 4, 4, 4, 4, 4};
  static constexpr int   mo[NP]  = {12, 4, 4, 12, 4, 4, 4, 4, 12, 4};
  static constexpr int   lt[NP]  = {0, 1, 2, 3, 4, 5, 2, 4, 3, 5};
  static constexpr int   ino[NP] = {0, 0, 12, 12, 12, 12, 24, 24, 24, 24};
  static constexpr int   oo[NP]  = {0, 12, 12, 0, 24, 12, 24, 12, 36, 24};
  static constexpr int   wf[NP]  = {0, 144, 192, 208, 256, 272, 288, 304, 320, 368};
  static constexpr float al[NP]  = {0.25f, 0.353553391f, 0.353553391f, 0.25f, 0.5f,
                                    0.353553391f, 0.5f, 0.353553391f, 0.5f, 0.5f};
};

// ---------- layer kernel: edge FC-MLP (LDS-staged weights) + tensor product + scatter ----------

template <int L>
__global__ __launch_bounds__(256) void layer_kernel(
    const float* __restrict__ node, const float* __restrict__ shbuf,
    const float* __restrict__ eabuf, const int* __restrict__ nbr,
    const float* __restrict__ w1, const float* __restrict__ b1,
    const float* __restrict__ w2, const float* __restrict__ b2,
    float* __restrict__ upd)
{
  using D = LayerDef<L>;
  constexpr int WN = D::WN;
  __shared__ float sW1[36 * 36];
  __shared__ float sB1[36];
  __shared__ float sW2[36 * WN];
  __shared__ float sB2[WN];
  for (int i = threadIdx.x; i < 36 * 36; i += 256) sW1[i] = w1[i];
  for (int i = threadIdx.x; i < 36;      i += 256) sB1[i] = b1[i];
  for (int i = threadIdx.x; i < 36 * WN; i += 256) sW2[i] = w2[i];
  for (int i = threadIdx.x; i < WN;      i += 256) sB2[i] = b2[i];
  __syncthreads();

  int e = blockIdx.x * blockDim.x + threadIdx.x;
  if (e >= NE) return;
  int eu = (e < EUND) ? e : e - EUND;
  int n0 = nbr[eu * 2 + 0];
  int n1 = nbr[eu * 2 + 1];
  int src = (e < EUND) ? n0 : n1;
  int dst = (e < EUND) ? n1 : n0;

  float sh[9];
#pragma unroll
  for (int i = 0; i < 9; ++i) sh[i] = shbuf[(size_t)e * 9 + i];

  float eall[36];
#pragma unroll
  for (int i = 0; i < 12; ++i) eall[i] = eabuf[(size_t)e * 12 + i];
#pragma unroll
  for (int i = 0; i < 12; ++i) eall[12 + i] = node[(size_t)src * 48 + i];
#pragma unroll
  for (int i = 0; i < 12; ++i) eall[24 + i] = node[(size_t)dst * 48 + i];

  float h[36];
#pragma unroll
  for (int i = 0; i < 36; ++i) {
    float v = sB1[i];
#pragma unroll
    for (int j = 0; j < 36; ++j) v += eall[j] * sW1[j * 36 + i];
    h[i] = fmaxf(v, 0.0f);
  }

  const float* xin = node + (size_t)dst * 48;
  constexpr int OUTD = D::OUTD;
  float out[OUTD];
#pragma unroll
  for (int k = 0; k < OUTD; ++k) out[k] = 0.0f;

  const float inv3 = 0.57735026919f;   // 1/sqrt(3)
  const float inv6 = 0.40824829046f;   // 1/sqrt(6)
  const float q121 = 0.31622776601f;   // 1/sqrt(10)
  const float r30  = 0.18257418584f;   // 1/sqrt(30)

#pragma unroll
  for (int p = 0; p < D::NP; ++p) {
    const int mi = D::mi[p], mo = D::mo[p], lt = D::lt[p];
    const int ino = D::ino[p], oo = D::oo[p], wf = D::wf[p];
    const float alpha = D::al[p];
    const int nk = (lt == 0 || lt == 3) ? 1 : 3;
    float acc[12];
#pragma unroll
    for (int t = 0; t < 12; ++t) acc[t] = 0.0f;

    for (int u = 0; u < mi; ++u) {
      float t0 = 0.0f, t1 = 0.0f, t2 = 0.0f;
      if (lt == 0) {
        t0 = xin[ino + u] * sh[0];
      } else if (lt == 1) {
        float xu = xin[ino + u];
        t0 = xu * sh[1] * inv3; t1 = xu * sh[2] * inv3; t2 = xu * sh[3] * inv3;
      } else {
        float x0 = xin[ino + 3 * u + 0];
        float x1 = xin[ino + 3 * u + 1];
        float x2 = xin[ino + 3 * u + 2];
        if (lt == 2) {
          t0 = x0 * inv3; t1 = x1 * inv3; t2 = x2 * inv3;            // * sh[0]==1
        } else if (lt == 3) {
          t0 = (x0 * sh[1] + x1 * sh[2] + x2 * sh[3]) * inv3;
        } else if (lt == 4) {
          t0 = (x1 * sh[3] - x2 * sh[2]) * inv6;
          t1 = (x2 * sh[1] - x0 * sh[3]) * inv6;
          t2 = (x0 * sh[2] - x1 * sh[1]) * inv6;
        } else {  // lt == 5 : (1,2,1) Wigner contraction
          t0 = q121 * (x1 * sh[5] + x2 * sh[4] - x0 * sh[8]) - r30 * x0 * sh[6];
          t1 = q121 * (x0 * sh[5] + x2 * sh[7]) + 2.0f * r30 * x1 * sh[6];
          t2 = q121 * (x0 * sh[4] + x1 * sh[7] + x2 * sh[8]) - r30 * x2 * sh[6];
        }
      }
#pragma unroll
      for (int w = 0; w < 12; ++w) {
        if (w < mo) {
          int cidx = wf + u * mo + w;
          float wp = sB2[cidx];
#pragma unroll
          for (int i = 0; i < 36; ++i) wp += h[i] * sW2[i * WN + cidx];
          if (nk == 1) {
            acc[w] += t0 * wp;
          } else {
            acc[3 * w + 0] += t0 * wp;
            acc[3 * w + 1] += t1 * wp;
            acc[3 * w + 2] += t2 * wp;
          }
        }
      }
    }
#pragma unroll
    for (int t = 0; t < 12; ++t)
      if (t < mo * nk) out[oo + t] += alpha * acc[t];
  }
#pragma unroll
  for (int k = 0; k < OUTD; ++k)
    atomicAdd(&upd[(size_t)src * 48 + k], out[k]);
}

// ---------- fused head MLP: WMMA GEMM(K=64) -> tanh -> LDS -> WMMA GEMM(K=512) ----------

__global__ __launch_bounds__(128) void mlp_head_kernel(
    const _Float16* __restrict__ nodeF16,
    const _Float16* __restrict__ pW1mu, const _Float16* __restrict__ pW1sg,
    const float* __restrict__ b1mu, const float* __restrict__ b1sg,
    const _Float16* __restrict__ pW2mu, const _Float16* __restrict__ pW2sg,
    const float* __restrict__ b2mu, const float* __restrict__ b2sg,
    float* __restrict__ out)
{
  __shared__ __align__(16) _Float16 hbuf[16 * 512];
  const int tile = blockIdx.x;
  const int head = blockIdx.y;
  const _Float16* pW1 = head ? pW1sg : pW1mu;
  const _Float16* pW2 = head ? pW2sg : pW2mu;
  const float* b1 = head ? b1sg : b1mu;
  const float* b2 = head ? b2sg : b2mu;
  const int wave = threadIdx.x >> 5;
  const int lane = threadIdx.x & 31;
  const int row0 = tile * 16;
  const int mb   = (lane >> 4) << 3;   // M base of C rows held by this lane
  const int ncl  = lane & 15;          // N column within tile

  // A fragments for K = 0..63 (node padded with zeros beyond 48)
  v16h a0 = load_a_frag(nodeF16 + (size_t)row0 * 64,      64, lane);
  v16h a1 = load_a_frag(nodeF16 + (size_t)row0 * 64 + 32, 64, lane);

  // GEMM1: hidden(16x512) = tanh(node @ W1 + b1), waves split the 32 n-tiles
  for (int nt = wave; nt < 32; nt += 4) {
    float bias = b1[nt * 16 + ncl];
    v8f c;
#pragma unroll
    for (int r = 0; r < 8; ++r) c[r] = bias;
    v16h bA = load_b_packed(pW1 + (size_t)(0 * 32 + nt) * 512, lane);
    c = __builtin_amdgcn_wmma_f32_16x16x32_f16(false, a0, false, bA, (short)0, c, false, false);
    v16h bB = load_b_packed(pW1 + (size_t)(1 * 32 + nt) * 512, lane);
    c = __builtin_amdgcn_wmma_f32_16x16x32_f16(false, a1, false, bB, (short)0, c, false, false);
#pragma unroll
    for (int r = 0; r < 8; ++r)
      hbuf[(mb + r) * 512 + nt * 16 + ncl] = (_Float16)tanhf(c[r]);
  }
  __syncthreads();

  // GEMM2: out(16x512) = hidden @ W2 + b2, K = 512 in 16 WMMA k-steps
  for (int nt = wave; nt < 32; nt += 4) {
    float bias = b2[nt * 16 + ncl];
    v8f c;
#pragma unroll
    for (int r = 0; r < 8; ++r) c[r] = bias;
    for (int kt = 0; kt < 16; ++kt) {
      v16h a = load_a_frag(hbuf + kt * 32, 512, lane);
      v16h b = load_b_packed(pW2 + (size_t)(kt * 32 + nt) * 512, lane);
      if (kt + 1 < 16)
        __builtin_prefetch(pW2 + (size_t)((kt + 1) * 32 + nt) * 512, 0, 1);
      c = __builtin_amdgcn_wmma_f32_16x16x32_f16(false, a, false, b, (short)0, c, false, false);
    }
#pragma unroll
    for (int r = 0; r < 8; ++r) {
      float v = c[r];
      if (head) v = 1e-9f + expf(0.5f * v);   // sigma head
      out[(size_t)head * NNODES * 512 + (size_t)(row0 + mb + r) * 512 + nt * 16 + ncl] = v;
    }
  }
}

// ---------- host launcher ----------
// Input order (pytree / alphabetical dict flatten):
// 0 cg_z 1 cg_xyz 2 cg_nbr_list, then params: 3 edge_b1 4 edge_b2 5 edge_w1 6 edge_w2,
// 7..18 fc[0..2]{b1,b2,w1,w2}, 19 mu_b1 20 mu_b2 21 mu_w1 22 mu_w2, 23 node_emb,
// 24 sig_b1 25 sig_b2 26 sig_w1 27 sig_w2

extern "C" void kernel_launch(void* const* d_in, const int* in_sizes, int n_in,
                              void* d_out, int out_size, void* d_ws, size_t ws_size,
                              hipStream_t stream)
{
  const int*   cg_z    = (const int*)d_in[0];
  const float* cg_xyz  = (const float*)d_in[1];
  const int*   nbr     = (const int*)d_in[2];
  const float* edge_b1 = (const float*)d_in[3];
  const float* edge_b2 = (const float*)d_in[4];
  const float* edge_w1 = (const float*)d_in[5];
  const float* edge_w2 = (const float*)d_in[6];
  const float* fcb1[3] = {(const float*)d_in[7],  (const float*)d_in[11], (const float*)d_in[15]};
  const float* fcb2[3] = {(const float*)d_in[8],  (const float*)d_in[12], (const float*)d_in[16]};
  const float* fcw1[3] = {(const float*)d_in[9],  (const float*)d_in[13], (const float*)d_in[17]};
  const float* fcw2[3] = {(const float*)d_in[10], (const float*)d_in[14], (const float*)d_in[18]};
  const float* mu_b1   = (const float*)d_in[19];
  const float* mu_b2   = (const float*)d_in[20];
  const float* mu_w1   = (const float*)d_in[21];
  const float* mu_w2   = (const float*)d_in[22];
  const float* node_emb= (const float*)d_in[23];
  const float* sig_b1  = (const float*)d_in[24];
  const float* sig_b2  = (const float*)d_in[25];
  const float* sig_w1  = (const float*)d_in[26];
  const float* sig_w2  = (const float*)d_in[27];

  char* wsp = (char*)d_ws;
  auto carve = [&](size_t bytes) -> char* {
    char* p = wsp;
    wsp += (bytes + 255) & ~(size_t)255;
    return p;
  };
  float*    shbuf   = (float*)carve((size_t)NE * 9 * 4);
  float*    eabuf   = (float*)carve((size_t)NE * 12 * 4);
  float*    counts  = (float*)carve((size_t)NNODES * 4);
  float*    node    = (float*)carve((size_t)NNODES * 48 * 4);
  float*    updb    = (float*)carve((size_t)NNODES * 48 * 4);
  _Float16* nodeF16 = (_Float16*)carve((size_t)NNODES * 64 * 2);
  _Float16* pW1mu   = (_Float16*)carve((size_t)64 * 512 * 2);
  _Float16* pW1sg   = (_Float16*)carve((size_t)64 * 512 * 2);
  _Float16* pW2mu   = (_Float16*)carve((size_t)512 * 512 * 2);
  _Float16* pW2sg   = (_Float16*)carve((size_t)512 * 512 * 2);

  hipMemsetAsync(counts, 0, (size_t)NNODES * 4, stream);

  edge_prep_kernel<<<(NE + 255) / 256, 256, 0, stream>>>(
      cg_z, cg_xyz, nbr, edge_w1, edge_b1, edge_w2, edge_b2, shbuf, eabuf, counts);
  node_init_kernel<<<(NNODES * 48 + 255) / 256, 256, 0, stream>>>(cg_z, node_emb, node);

  hipMemsetAsync(updb, 0, (size_t)NNODES * 48 * 4, stream);
  layer_kernel<0><<<(NE + 255) / 256, 256, 0, stream>>>(
      node, shbuf, eabuf, nbr, fcw1[0], fcb1[0], fcw2[0], fcb2[0], updb);
  node_update_kernel<<<(NNODES * 48 + 255) / 256, 256, 0, stream>>>(updb, counts, node);

  hipMemsetAsync(updb, 0, (size_t)NNODES * 48 * 4, stream);
  layer_kernel<1><<<(NE + 255) / 256, 256, 0, stream>>>(
      node, shbuf, eabuf, nbr, fcw1[1], fcb1[1], fcw2[1], fcb2[1], updb);
  node_update_kernel<<<(NNODES * 48 + 255) / 256, 256, 0, stream>>>(updb, counts, node);

  hipMemsetAsync(updb, 0, (size_t)NNODES * 48 * 4, stream);
  layer_kernel<2><<<(NE + 255) / 256, 256, 0, stream>>>(
      node, shbuf, eabuf, nbr, fcw1[2], fcb1[2], fcw2[2], fcb2[2], updb);
  node_update_kernel<<<(NNODES * 48 + 255) / 256, 256, 0, stream>>>(updb, counts, node);

  node_to_f16_kernel<<<(NNODES * 64 + 255) / 256, 256, 0, stream>>>(node, nodeF16);

  pack_b_kernel<<<(2 * 32 * 512 + 255) / 256, 256, 0, stream>>>(mu_w1,  pW1mu, 48, 512, 64);
  pack_b_kernel<<<(2 * 32 * 512 + 255) / 256, 256, 0, stream>>>(sig_w1, pW1sg, 48, 512, 64);
  pack_b_kernel<<<(16 * 32 * 512 + 255) / 256, 256, 0, stream>>>(mu_w2,  pW2mu, 512, 512, 512);
  pack_b_kernel<<<(16 * 32 * 512 + 255) / 256, 256, 0, stream>>>(sig_w2, pW2sg, 512, 512, 512);

  mlp_head_kernel<<<dim3(NNODES / 16, 2), 128, 0, stream>>>(
      nodeF16, pW1mu, pW1sg, mu_b1, sig_b1, pW2mu, pW2sg, mu_b2, sig_b2, (float*)d_out);

  (void)in_sizes; (void)n_in; (void)out_size; (void)ws_size;
}
